// JSDMAttention_24034636988442
// MI455X (gfx1250) — matrix operand order repaired and checked by reading, not compiled, not run on hardware
//
#include <hip/hip_runtime.h>

// ---------------- problem constants ----------------
constexpr int kB  = 4;
constexpr int kS  = 100;
constexpr int kT  = 128;
constexpr int kC  = 256;
constexpr int kH  = 256;
constexpr int kNH = 4;
constexpr int kDH = 32;
constexpr int kAH = 128;   // NH*DH
constexpr int kFH = 32;
constexpr float kMaxSp  = 180.0f;
constexpr float kMaxTm  = 365.0f;
constexpr float kMaxDoy = 182.0f;

typedef __attribute__((ext_vector_type(16))) __bf16          v16bf;
typedef __attribute__((ext_vector_type(16))) unsigned short  v16us;
typedef __attribute__((ext_vector_type(8)))  unsigned short  v8us;
typedef __attribute__((ext_vector_type(8)))  float           v8f;

union Frag { v16bf b; v16us u; };

__device__ __forceinline__ unsigned short f2bf(float f) {
  unsigned int u = __float_as_uint(f);
  u += 0x7fffu + ((u >> 16) & 1u);           // round-to-nearest-even
  return (unsigned short)(u >> 16);
}

__device__ __forceinline__ Frag load_frag(const unsigned short* plo,
                                          const unsigned short* phi) {
  v8us lo = *(const v8us*)plo;
  v8us hi = *(const v8us*)phi;
  Frag f;
  f.u = __builtin_shufflevector(lo, hi, 0,1,2,3,4,5,6,7,8,9,10,11,12,13,14,15);
  return f;
}

__device__ __forceinline__ v8f wmma_bf16(Frag a, Frag b, v8f c) {
  // D = A(16x32 bf16) * B(32x16 bf16) + C(16x16 f32)
  return __builtin_amdgcn_wmma_f32_16x16x32_bf16(false, a.b, false, b.b,
                                                 (short)0, c, false, false);
}

// ---------------- fp32 -> bf16 conversion (weights) ----------------
__global__ void cvt_bf16_kernel(const float* __restrict__ src,
                                unsigned short* __restrict__ dst, int n) {
  int i = blockIdx.x * blockDim.x + threadIdx.x;
  if (i < n) dst[i] = f2bf(src[i]);
}

// ---------------- FIRE distance bias (mask folded into db) ----------------
__device__ __forceinline__ float fire_mlp(float dist, float c,
                                          const float* __restrict__ w1,
                                          const float* __restrict__ w2,
                                          float maxd) {
  c = fmaxf(c, 0.0f);
  float d = __logf(c * dist + 1.0f) / __logf(c * maxd + 1.0f);
  float out = 0.0f;
#pragma unroll
  for (int j = 0; j < kFH; ++j) {
    float h  = d * w1[j];
    float sl = h / (1.0f + __expf(-h));      // SiLU
    out += sl * w2[j];
  }
  return out;
}

__global__ void fire_kernel(const float* __restrict__ st,
                            const float* __restrict__ doy,
                            const float* __restrict__ mask,
                            const float* __restrict__ cs, const float* __restrict__ w1s, const float* __restrict__ w2s,
                            const float* __restrict__ ctm, const float* __restrict__ w1t, const float* __restrict__ w2t,
                            const float* __restrict__ cdy, const float* __restrict__ w1d, const float* __restrict__ w2d,
                            float* __restrict__ sb, float* __restrict__ tb, float* __restrict__ db) {
  int i = blockIdx.x * blockDim.x + threadIdx.x;
  if (i >= kB * kT * kC) return;
  float sp = st[2 * i + 0];
  float tm = st[2 * i + 1];
  float dy = doy[i];
  sb[i] = fire_mlp(sp, *cs,  w1s, w2s, kMaxSp);
  tb[i] = fire_mlp(tm, *ctm, w1t, w2t, kMaxTm);
  db[i] = fire_mlp(dy, *cdy, w1d, w2d, kMaxDoy) + mask[i];  // mask has same (b,t,c) index
}

// ---------------- generic WMMA GEMM:  Y(M,N) = A(M,K) * W(N,K)^T + bias ----------------
// Block: 256 threads = 8 waves; each block does 128 rows x all N.
// W (N x K bf16) staged once per block into LDS with +8 row padding (bank skew).
template <int N, int K, bool ABF16, bool OUTF32>
__global__ __launch_bounds__(256) void gemm_wmma_kernel(
    const float* __restrict__ Af, const unsigned short* __restrict__ Ab,
    const unsigned short* __restrict__ Wb, const float* __restrict__ bias,
    float* __restrict__ Yf, unsigned short* __restrict__ Yb) {
  constexpr int NT = N / 16;
  constexpr int KP = K + 8;           // padded LDS row stride (ushorts)
  constexpr int CH = K / 8;           // v8us chunks per W row
  extern __shared__ unsigned short wl[];

  // cooperative W staging (coalesced b128 loads, padded LDS rows)
  for (int i = threadIdx.x; i < N * CH; i += 256) {
    const int row = i / CH, c8 = i % CH;
    *(v8us*)(wl + row * KP + c8 * 8) = *(const v8us*)(Wb + (size_t)row * K + c8 * 8);
  }
  __syncthreads();

  const int lane = threadIdx.x & 31;
  const int w    = threadIdx.x >> 5;
  const int l15  = lane & 15;
  const int m0   = blockIdx.x * 128 + w * 16;
  const int offA = (lane < 16) ? 0 : 8;    // A per-lane K pattern {0..7,16..23}/{8..15,24..31}
  const int offB = (lane < 16) ? 0 : 8;    // B per-lane K pattern {0..7,16..23}/{8..15,24..31}

  v8f acc[NT];
#pragma unroll
  for (int nt = 0; nt < NT; ++nt)
#pragma unroll
    for (int j = 0; j < 8; ++j) acc[nt][j] = 0.0f;

  const int rA = m0 + l15;
  for (int k0 = 0; k0 < K; k0 += 32) {
    Frag fa;
    if (ABF16) {
      const unsigned short* ap = Ab + (size_t)rA * K + k0 + offA;
      fa = load_frag(ap, ap + 16);
    } else {
      const float* ap = Af + (size_t)rA * K + k0 + offA;
      float4 x0 = *(const float4*)(ap + 0);
      float4 x1 = *(const float4*)(ap + 4);
      float4 x2 = *(const float4*)(ap + 16);
      float4 x3 = *(const float4*)(ap + 20);
      fa.u[0]  = f2bf(x0.x); fa.u[1]  = f2bf(x0.y); fa.u[2]  = f2bf(x0.z); fa.u[3]  = f2bf(x0.w);
      fa.u[4]  = f2bf(x1.x); fa.u[5]  = f2bf(x1.y); fa.u[6]  = f2bf(x1.z); fa.u[7]  = f2bf(x1.w);
      fa.u[8]  = f2bf(x2.x); fa.u[9]  = f2bf(x2.y); fa.u[10] = f2bf(x2.z); fa.u[11] = f2bf(x2.w);
      fa.u[12] = f2bf(x3.x); fa.u[13] = f2bf(x3.y); fa.u[14] = f2bf(x3.z); fa.u[15] = f2bf(x3.w);
    }
#pragma unroll
    for (int nt = 0; nt < NT; ++nt) {
      const unsigned short* bp = wl + (nt * 16 + l15) * KP + k0 + offB;
      Frag fb = load_frag(bp, bp + 16);
      acc[nt] = wmma_bf16(fa, fb, acc[nt]);
    }
  }

  const int rbase = m0 + ((lane < 16) ? 0 : 8);
#pragma unroll
  for (int nt = 0; nt < NT; ++nt) {
    const int col = nt * 16 + l15;
    const float bv = bias[col];
#pragma unroll
    for (int i = 0; i < 8; ++i) {
      size_t idx = (size_t)(rbase + i) * N + col;
      float y = acc[nt][i] + bv;
      if (OUTF32) Yf[idx] = y; else Yb[idx] = f2bf(y);
    }
  }
}

// ---------------- fused attention: scores + bias + softmax + probs@V ----------------
// One block per (b, s, head). 8 waves; wave w owns t-rows [16w, 16w+16).
// LDS (padded rows, conflict-free ds_load_b128 fragments):
//   probs : kT  x (kC+8)  bf16
//   vT    : kDH x (kC+8)  bf16  (V transposed, d-major)
//   kL    : kC  x (kDH+8) bf16  (K slice, staged once)
__global__ __launch_bounds__(256) void attn_kernel(
    const unsigned short* __restrict__ qw, const unsigned short* __restrict__ kw,
    const unsigned short* __restrict__ vw,
    const float* __restrict__ sb, const float* __restrict__ tbv, const float* __restrict__ db,
    const float* __restrict__ sls, const float* __restrict__ tls,
    unsigned short* __restrict__ ctx) {
  constexpr int PP  = kC + 8;    // probs row stride
  constexpr int VP  = kC + 8;    // vT row stride
  constexpr int KLP = kDH + 8;   // kL row stride
  extern __shared__ unsigned short smem[];
  unsigned short* probs = smem;                     // kT*PP
  unsigned short* vT    = probs + kT * PP;          // kDH*VP
  unsigned short* kL    = vT + kDH * VP;            // kC*KLP

  const int blk = blockIdx.x;
  const int h = blk % kNH;
  const int s = (blk / kNH) % kS;
  const int b = blk / (kNH * kS);
  const int tid = threadIdx.x, lane = tid & 31, w = tid >> 5, l15 = lane & 15;

  const size_t slice_q  = (((size_t)b * kS + s) * kT) * kAH + h * kDH;  // + t*kAH
  const size_t slice_kv = (((size_t)b * kS + s) * kC) * kAH + h * kDH;  // + c*kAH

  // stage V^T (d-major) and K slice into LDS
  {
    const int c = tid;  // 0..255 == kC
    const v8us* vp = (const v8us*)(vw + slice_kv + (size_t)c * kAH);
    v8us r0 = vp[0], r1 = vp[1], r2 = vp[2], r3 = vp[3];
#pragma unroll
    for (int j = 0; j < 8; ++j) {
      vT[(0  + j) * VP + c] = r0[j];
      vT[(8  + j) * VP + c] = r1[j];
      vT[(16 + j) * VP + c] = r2[j];
      vT[(24 + j) * VP + c] = r3[j];
    }
    const v8us* kp = (const v8us*)(kw + slice_kv + (size_t)c * kAH);
#pragma unroll
    for (int j = 0; j < 4; ++j)
      *(v8us*)(kL + c * KLP + j * 8) = kp[j];
  }

  const int m0   = w * 16;
  const int offA = (lane < 16) ? 0 : 8;
  const int offB = (lane < 16) ? 0 : 8;
  const int thalf = (lane < 16) ? 0 : 8;

  // q fragment (A: 16 x 32, K = DH = 32 -> exactly one WMMA K-step)
  Frag fq;
  {
    const unsigned short* qp = qw + slice_q + (size_t)(m0 + l15) * kAH + offA;
    fq = load_frag(qp, qp + 16);
  }

  __syncthreads();  // vT + kL staged

  v8f acc[16];  // 16 c-tiles of 16x16 scores
#pragma unroll
  for (int n = 0; n < 16; ++n)
#pragma unroll
    for (int j = 0; j < 8; ++j) acc[n][j] = 0.0f;

#pragma unroll
  for (int cti = 0; cti < 16; ++cti) {
    const unsigned short* kp = kL + (cti * 16 + l15) * KLP + offB;
    Frag fk = load_frag(kp, kp + 16);
    acc[cti] = wmma_bf16(fq, fk, acc[cti]);
  }

  // fuse scale + FIRE bias (mask already folded into db) in the C/D layout
  const float es = __expf(sls[s]);
  const float et = __expf(tls[s]);
  const float inv = 0.17677669529663688f;  // 1/sqrt(32)
#pragma unroll
  for (int cti = 0; cti < 16; ++cti) {
    const int c = cti * 16 + l15;
#pragma unroll
    for (int i = 0; i < 8; ++i) {
      const int t = m0 + i + thalf;
      const size_t bi = ((size_t)b * kT + t) * kC + c;
      acc[cti][i] = acc[cti][i] * inv + sb[bi] * es + tbv[bi] * et + db[bi];
    }
  }

  // register softmax over c (16 tiles in-lane + 16-lane half reduction)
#pragma unroll
  for (int i = 0; i < 8; ++i) {
    float mx = acc[0][i];
#pragma unroll
    for (int cti = 1; cti < 16; ++cti) mx = fmaxf(mx, acc[cti][i]);
#pragma unroll
    for (int o = 8; o >= 1; o >>= 1) mx = fmaxf(mx, __shfl_xor(mx, o, 32));
    float sum = 0.0f;
#pragma unroll
    for (int cti = 0; cti < 16; ++cti) {
      float e = __expf(acc[cti][i] - mx);
      acc[cti][i] = e;
      sum += e;
    }
#pragma unroll
    for (int o = 8; o >= 1; o >>= 1) sum += __shfl_xor(sum, o, 32);
    const float rs = 1.0f / sum;
    const int t = m0 + i + thalf;
#pragma unroll
    for (int cti = 0; cti < 16; ++cti)
      probs[t * PP + cti * 16 + l15] = f2bf(acc[cti][i] * rs);
  }

  __syncthreads();

  // ctx = probs(128x256) @ V(256x32); per wave: 2 n-tiles, 8 K-steps
  v8f o0, o1;
#pragma unroll
  for (int j = 0; j < 8; ++j) { o0[j] = 0.0f; o1[j] = 0.0f; }
  const int tA = m0 + l15;
#pragma unroll
  for (int c0 = 0; c0 < kC; c0 += 32) {
    const unsigned short* pp = probs + tA * PP + c0 + offA;
    Frag fp = load_frag(pp, pp + 16);
    const unsigned short* v0p = vT + (0 + l15) * VP + c0 + offB;
    Frag f0 = load_frag(v0p, v0p + 16);
    o0 = wmma_bf16(fp, f0, o0);
    const unsigned short* v1p = vT + (16 + l15) * VP + c0 + offB;
    Frag f1 = load_frag(v1p, v1p + 16);
    o1 = wmma_bf16(fp, f1, o1);
  }

#pragma unroll
  for (int i = 0; i < 8; ++i) {
    const int t = m0 + i + thalf;
    const size_t base = slice_q + (size_t)t * kAH;
    ctx[base + l15]      = f2bf(o0[i]);
    ctx[base + 16 + l15] = f2bf(o1[i]);
  }
}

// ---------------- launcher ----------------
extern "C" void kernel_launch(void* const* d_in, const int* in_sizes, int n_in,
                              void* d_out, int out_size, void* d_ws, size_t ws_size,
                              hipStream_t stream) {
  (void)in_sizes; (void)n_in; (void)out_size; (void)ws_size;

  const float* hs   = (const float*)d_in[0];
  const float* src  = (const float*)d_in[1];
  const float* mask = (const float*)d_in[2];
  const float* st   = (const float*)d_in[3];
  const float* doy  = (const float*)d_in[4];
  const float* Wq = (const float*)d_in[5];  const float* bq = (const float*)d_in[6];
  const float* Wk = (const float*)d_in[7];  const float* bk = (const float*)d_in[8];
  const float* Wv = (const float*)d_in[9];  const float* bv = (const float*)d_in[10];
  const float* Wo = (const float*)d_in[11]; const float* bo = (const float*)d_in[12];
  const float* cs  = (const float*)d_in[13]; const float* w1s = (const float*)d_in[14]; const float* w2s = (const float*)d_in[15];
  const float* ctm = (const float*)d_in[16]; const float* w1t = (const float*)d_in[17]; const float* w2t = (const float*)d_in[18];
  const float* cdy = (const float*)d_in[19]; const float* w1d = (const float*)d_in[20]; const float* w2d = (const float*)d_in[21];
  const float* sls = (const float*)d_in[22]; const float* tls = (const float*)d_in[23];

  // workspace layout (~80.5 MB total)
  const size_t nQ    = (size_t)kB * kS * kT * kAH;  // 6,553,600
  const size_t nKV   = (size_t)kB * kS * kC * kAH;  // 13,107,200
  const size_t nBias = (size_t)kB * kT * kC;        // 131,072
  const size_t nW    = (size_t)kAH * kH;            // 32,768

  char* p = (char*)d_ws;
  unsigned short* qw  = (unsigned short*)p; p += nQ  * 2;
  unsigned short* kw  = (unsigned short*)p; p += nKV * 2;
  unsigned short* vw  = (unsigned short*)p; p += nKV * 2;
  unsigned short* cw  = (unsigned short*)p; p += nQ  * 2;
  float* sbw = (float*)p; p += nBias * 4;
  float* tbw = (float*)p; p += nBias * 4;
  float* dbw = (float*)p; p += nBias * 4;
  unsigned short* wqb = (unsigned short*)p; p += nW * 2;
  unsigned short* wkb = (unsigned short*)p; p += nW * 2;
  unsigned short* wvb = (unsigned short*)p; p += nW * 2;
  unsigned short* wob = (unsigned short*)p; p += nW * 2;

  // 1) weights -> bf16
  const int cvtGrid = (int)((nW + 255) / 256);
  cvt_bf16_kernel<<<cvtGrid, 256, 0, stream>>>(Wq, wqb, (int)nW);
  cvt_bf16_kernel<<<cvtGrid, 256, 0, stream>>>(Wk, wkb, (int)nW);
  cvt_bf16_kernel<<<cvtGrid, 256, 0, stream>>>(Wv, wvb, (int)nW);
  cvt_bf16_kernel<<<cvtGrid, 256, 0, stream>>>(Wo, wob, (int)nW);

  // 2) FIRE biases over (B,T,C); attention_mask folded into db
  fire_kernel<<<(int)((nBias + 255) / 256), 256, 0, stream>>>(
      st, doy, mask, cs, w1s, w2s, ctm, w1t, w2t, cdy, w1d, w2d, sbw, tbw, dbw);

  // 3) Q/K/V projections (bf16 WMMA, fp32 accumulate, bf16 out), W LDS-resident
  const int smemQKV = kAH * (kH + 8) * 2;   // 67,584 B
  gemm_wmma_kernel<kAH, kH, false, false><<<400, 256, smemQKV, stream>>>(
      hs,  nullptr, wqb, bq, nullptr, qw);   // M = B*S*T = 51200
  gemm_wmma_kernel<kAH, kH, false, false><<<800, 256, smemQKV, stream>>>(
      src, nullptr, wkb, bk, nullptr, kw);   // M = B*S*C = 102400
  gemm_wmma_kernel<kAH, kH, false, false><<<800, 256, smemQKV, stream>>>(
      src, nullptr, wvb, bv, nullptr, vw);

  // 4) fused attention; ~102.5 KB dynamic LDS (probs + V^T + K slice, padded)
  const int attnSmem = (kT * (kC + 8) + kDH * (kC + 8) + kC * (kDH + 8)) * 2;
  attn_kernel<<<kB * kS * kNH, 256, attnSmem, stream>>>(
      qw, kw, vw, sbw, tbw, dbw, sls, tls, cw);

  // 5) output dense -> fp32 d_out
  const int smemO = kH * (kAH + 8) * 2;     // 69,632 B
  gemm_wmma_kernel<kH, kAH, true, true><<<400, 256, smemO, stream>>>(
      nullptr, cw, wob, bo, (float*)d_out, nullptr);
}
